// MoleculeOrganismInteraction_32753420599330
// MI455X (gfx1250) — matrix operand order repaired and checked by reading, not compiled, not run on hardware
//
#include <hip/hip_runtime.h>

// ---------------------------------------------------------------------------
// Problem constants (from reference): FD=1024, FO=512, FK=128, H=4, N=32768
// ---------------------------------------------------------------------------
#define N_ROWS 32768
#define FD 1024
#define FO 512
#define FKH 512          // FK*H
#define FDH 4096         // FD*H

typedef __attribute__((ext_vector_type(16))) __bf16          v16bf;
typedef __attribute__((ext_vector_type(8)))  float           v8f;
typedef __attribute__((ext_vector_type(8)))  unsigned short  u16x8;

union Frag16 {
    u16x8 h2[2];
    v16bf v;
};

// round-to-nearest-even f32 -> bf16 (inputs are finite normals; skip NaN path)
static __device__ __forceinline__ unsigned short f2bf(float f) {
    unsigned int u = __float_as_uint(f);
    u += 0x7FFFu + ((u >> 16) & 1u);
    return (unsigned short)(u >> 16);
}

// ---------------------------------------------------------------------------
// 1) f32 -> bf16 streaming convert (vectorized: float4 x2 in, 16B out)
// ---------------------------------------------------------------------------
__global__ __launch_bounds__(256) void conv_bf16_kernel(
    const float* __restrict__ in, unsigned short* __restrict__ out, int n8)
{
    int i = blockIdx.x * blockDim.x + threadIdx.x;
    int stride = gridDim.x * blockDim.x;
    for (; i < n8; i += stride) {
        const float4* p = (const float4*)in + (size_t)i * 2;
        float4 a = p[0];
        float4 b = p[1];
        u16x8 r;
        r[0] = f2bf(a.x); r[1] = f2bf(a.y); r[2] = f2bf(a.z); r[3] = f2bf(a.w);
        r[4] = f2bf(b.x); r[5] = f2bf(b.y); r[6] = f2bf(b.z); r[7] = f2bf(b.w);
        *((u16x8*)out + i) = r;
    }
}

// ---------------------------------------------------------------------------
// 2) transpose + convert: in (K x C, f32 row-major) -> out (C x K, bf16)
//    K, C multiples of 32. LDS-tiled, coalesced both ways.
// ---------------------------------------------------------------------------
__global__ __launch_bounds__(256) void transpose_bf16_kernel(
    const float* __restrict__ in, unsigned short* __restrict__ out, int K, int C)
{
    __shared__ float tile[32][33];
    const int tx = threadIdx.x, ty = threadIdx.y;
    const int c0 = blockIdx.x * 32, k0 = blockIdx.y * 32;
#pragma unroll
    for (int j = 0; j < 32; j += 8)
        tile[ty + j][tx] = in[(size_t)(k0 + ty + j) * C + (c0 + tx)];
    __syncthreads();
#pragma unroll
    for (int j = 0; j < 32; j += 8)
        out[(size_t)(c0 + ty + j) * K + (k0 + tx)] = f2bf(tile[tx][ty + j]);
}

// ---------------------------------------------------------------------------
// Fragment loaders (bf16 WMMA 16x16x32, wave32):
//  A (16xK tile): lane l: row = l&15;
//    lanes 0-15 hold K = k0..k0+7 and k0+16..k0+23 ; lanes 16-31 offset by 8.
//  B (Kx16 tile): lane l: col = l&15, same K order as A:
//    lanes 0-15 hold K = k0..k0+7 and k0+16..k0+23 ; lanes 16-31 offset by 8.
//  With row-major bf16 activations and column-major (transposed) bf16
//  weights both fragments are plain aligned b128 loads.
// ---------------------------------------------------------------------------
static __device__ __forceinline__ Frag16 loadA(
    const unsigned short* __restrict__ base, int half)
{
    Frag16 f;
    const unsigned short* p = base + half * 8;
    f.h2[0] = *(const u16x8*)(p);
    f.h2[1] = *(const u16x8*)(p + 16);
    return f;
}

static __device__ __forceinline__ Frag16 loadB(
    const unsigned short* __restrict__ base, int half)
{
    Frag16 f;
    const unsigned short* p = base + half * 8;
    f.h2[0] = *(const u16x8*)(p);
    f.h2[1] = *(const u16x8*)(p + 16);
    return f;
}

// ---------------------------------------------------------------------------
// 3) Fused Q/K GEMM + per-head dot -> scores (N x 4)
//    Block = 256 threads = 8 waves, handles 16 rows.
//    Wave w owns 64 columns [64w, 64w+64) of the 512-wide Q/K space
//    (entirely inside head h = w>>1). Q,K tiles stay in registers.
// ---------------------------------------------------------------------------
__global__ __launch_bounds__(256) void qk_scores_kernel(
    const unsigned short* __restrict__ Xo_bf,  // N x 512
    const unsigned short* __restrict__ Xd_bf,  // N x 1024
    const unsigned short* __restrict__ WqT,    // 512 x 512   [c][k]
    const unsigned short* __restrict__ WkT,    // 512 x 1024  [c][k]
    const float* __restrict__ bq,              // 512
    const float* __restrict__ bk,              // 512
    float* __restrict__ scores)                // N x 4
{
    __shared__ float sblk[16][8];
    const int wave = threadIdx.x >> 5;
    const int lane = threadIdx.x & 31;
    const int half = lane >> 4;
    const int l16  = lane & 15;
    const int m0   = blockIdx.x * 16;
    const int cb   = wave * 64;

    v8f qacc[4] = {};
    v8f kacc[4] = {};

    // Q = Xo @ Wq : reduction K = 512
    {
        const unsigned short* paB = Xo_bf + (size_t)(m0 + l16) * FO;
        for (int k0 = 0; k0 < FO; k0 += 32) {
            Frag16 A = loadA(paB + k0, half);
#pragma unroll
            for (int t = 0; t < 4; ++t) {
                const unsigned short* pb = WqT + (size_t)(cb + 16 * t + l16) * FO + k0;
                Frag16 B = loadB(pb, half);
                qacc[t] = __builtin_amdgcn_wmma_f32_16x16x32_bf16(
                    false, A.v, false, B.v, (short)0, qacc[t], false, false);
            }
        }
    }
    // K = Xd @ Wk : reduction K = 1024
    {
        const unsigned short* paB = Xd_bf + (size_t)(m0 + l16) * FD;
        for (int k0 = 0; k0 < FD; k0 += 32) {
            Frag16 A = loadA(paB + k0, half);
#pragma unroll
            for (int t = 0; t < 4; ++t) {
                const unsigned short* pb = WkT + (size_t)(cb + 16 * t + l16) * FD + k0;
                Frag16 B = loadB(pb, half);
                kacc[t] = __builtin_amdgcn_wmma_f32_16x16x32_bf16(
                    false, A.v, false, B.v, (short)0, kacc[t], false, false);
            }
        }
    }

    // C/D layout: VGPR j, lane l -> row = j + 8*(l>>4), col = l&15.
    // Per-row dot over this wave's 64 columns: sum over t, then over 16 lanes.
    float jsum[8];
#pragma unroll
    for (int j = 0; j < 8; ++j) jsum[j] = 0.0f;
#pragma unroll
    for (int t = 0; t < 4; ++t) {
        const int ci = cb + 16 * t + l16;
        const float bqv = bq[ci];
        const float bkv = bk[ci];
#pragma unroll
        for (int j = 0; j < 8; ++j)
            jsum[j] += (qacc[t][j] + bqv) * (kacc[t][j] + bkv);
    }
#pragma unroll
    for (int j = 0; j < 8; ++j) {
        float s = jsum[j];
        s += __shfl_xor(s, 1, 32);
        s += __shfl_xor(s, 2, 32);
        s += __shfl_xor(s, 4, 32);
        s += __shfl_xor(s, 8, 32);   // stays within the 16-lane column group
        jsum[j] = s;
    }
    if (l16 == 0) {
#pragma unroll
        for (int j = 0; j < 8; ++j)
            sblk[j + 8 * half][wave] = jsum[j];
    }
    __syncthreads();
    if (threadIdx.x < 64) {
        const int r = threadIdx.x >> 2, h = threadIdx.x & 3;
        scores[(size_t)(m0 + r) * 4 + h] = sblk[r][2 * h] + sblk[r][2 * h + 1];
    }
}

// ---------------------------------------------------------------------------
// 4) Fused V GEMM + score-scale + head-max -> out (N x 1024)
//    Grid (N/32, 1024/128). Block = 8 waves, all along N.
//    Each wave: 32 rows x 16 cols x 4 heads -> 8 accumulators; every B
//    fragment feeds 2 WMMAs (12 b128 loads per 8 WMMAs = 1.5 loads/WMMA).
//    The 512 MB V intermediate never exists in memory.
// ---------------------------------------------------------------------------
__global__ __launch_bounds__(256) void v_out_kernel(
    const unsigned short* __restrict__ Xd_bf,   // N x 1024
    const unsigned short* __restrict__ WvT,     // 4096 x 1024  [col][k]
    const float* __restrict__ bv,               // 4096
    const float* __restrict__ scores,           // N x 4
    float* __restrict__ out)                    // N x 1024
{
    __shared__ float ssc[32][4];
    const int wave = threadIdx.x >> 5;
    const int lane = threadIdx.x & 31;
    const int half = lane >> 4;
    const int l16  = lane & 15;
    const int rowBlk = blockIdx.x * 32;
    const int col = blockIdx.y * 128 + wave * 16 + l16;

    if (threadIdx.x < 128) {
        const int r = threadIdx.x >> 2, h = threadIdx.x & 3;
        ssc[r][h] = scores[(size_t)(rowBlk + r) * 4 + h];
    }
    __syncthreads();

    v8f acc[4][2] = {};
    const unsigned short* paB0 = Xd_bf + (size_t)(rowBlk + l16) * FD;       // rows +0..15
    const unsigned short* paB1 = Xd_bf + (size_t)(rowBlk + 16 + l16) * FD;  // rows +16..31

    for (int k0 = 0; k0 < FD; k0 += 32) {
        Frag16 A0 = loadA(paB0 + k0, half);
        Frag16 A1 = loadA(paB1 + k0, half);
        if (k0 + 32 < FD) {
            __builtin_prefetch(paB0 + k0 + 32, 0, 1);  // global_prefetch next A tiles
            __builtin_prefetch(paB1 + k0 + 32, 0, 1);
        }
#pragma unroll
        for (int h = 0; h < 4; ++h) {
            const unsigned short* pb = WvT + (size_t)(h * 1024 + col) * FD + k0;
            Frag16 B = loadB(pb, half);
            acc[h][0] = __builtin_amdgcn_wmma_f32_16x16x32_bf16(
                false, A0.v, false, B.v, (short)0, acc[h][0], false, false);
            acc[h][1] = __builtin_amdgcn_wmma_f32_16x16x32_bf16(
                false, A1.v, false, B.v, (short)0, acc[h][1], false, false);
        }
    }

    float bvv[4];
#pragma unroll
    for (int h = 0; h < 4; ++h) bvv[h] = bv[h * 1024 + col];

#pragma unroll
    for (int m = 0; m < 2; ++m) {
#pragma unroll
        for (int j = 0; j < 8; ++j) {
            const int r = m * 16 + j + 8 * half;   // row within the 32-row block
            float v0 = ssc[r][0] * (acc[0][m][j] + bvv[0]);
            float v1 = ssc[r][1] * (acc[1][m][j] + bvv[1]);
            float v2 = ssc[r][2] * (acc[2][m][j] + bvv[2]);
            float v3 = ssc[r][3] * (acc[3][m][j] + bvv[3]);
            out[(size_t)(rowBlk + r) * FD + col] =
                fmaxf(fmaxf(v0, v1), fmaxf(v2, v3));
        }
    }
}

// ---------------------------------------------------------------------------
// Launch
// ---------------------------------------------------------------------------
extern "C" void kernel_launch(void* const* d_in, const int* in_sizes, int n_in,
                              void* d_out, int out_size, void* d_ws, size_t ws_size,
                              hipStream_t stream) {
    const float* Xd = (const float*)d_in[0];   // N x 1024
    const float* Xo = (const float*)d_in[1];   // N x 512
    const float* Wq = (const float*)d_in[2];   // 512 x 512
    const float* bq = (const float*)d_in[3];   // 512
    const float* Wk = (const float*)d_in[4];   // 1024 x 512
    const float* bk = (const float*)d_in[5];   // 512
    const float* Wv = (const float*)d_in[6];   // 1024 x 4096
    const float* bv = (const float*)d_in[7];   // 4096
    float* out = (float*)d_out;

    // Workspace layout (total ~111 MB)
    char* ws = (char*)d_ws;
    size_t off = 0;
    unsigned short* Xd_bf = (unsigned short*)(ws + off); off += (size_t)N_ROWS * FD * 2;  // 64 MB
    unsigned short* Xo_bf = (unsigned short*)(ws + off); off += (size_t)N_ROWS * FO * 2;  // 32 MB
    unsigned short* WqT   = (unsigned short*)(ws + off); off += (size_t)FKH * FO * 2;     // 0.5 MB
    unsigned short* WkT   = (unsigned short*)(ws + off); off += (size_t)FKH * FD * 2;     // 1 MB
    unsigned short* WvT   = (unsigned short*)(ws + off); off += (size_t)FDH * FD * 2;     // 8 MB
    float*          sc    = (float*)(ws + off);          off += (size_t)N_ROWS * 4 * 4;   // 0.5 MB
    (void)ws_size; (void)in_sizes; (void)n_in; (void)out_size;

    // 1) convert activations to bf16
    conv_bf16_kernel<<<8192, 256, 0, stream>>>(Xd, Xd_bf, (N_ROWS * FD) / 8);
    conv_bf16_kernel<<<4096, 256, 0, stream>>>(Xo, Xo_bf, (N_ROWS * FO) / 8);

    // 2) transpose-convert weights (K-major -> col-major so B fragments are
    //    contiguous b128 loads)
    {
        dim3 blk(32, 8);
        transpose_bf16_kernel<<<dim3(FKH / 32, FO / 32), blk, 0, stream>>>(Wq, WqT, FO, FKH);
        transpose_bf16_kernel<<<dim3(FKH / 32, FD / 32), blk, 0, stream>>>(Wk, WkT, FD, FKH);
        transpose_bf16_kernel<<<dim3(FDH / 32, FD / 32), blk, 0, stream>>>(Wv, WvT, FD, FDH);
    }

    // 3) fused Q/K GEMMs + per-head dot -> scores
    qk_scores_kernel<<<N_ROWS / 16, 256, 0, stream>>>(Xo_bf, Xd_bf, WqT, WkT, bq, bk, sc);

    // 4) fused V GEMM + score scaling + head max -> out
    v_out_kernel<<<dim3(N_ROWS / 32, FD / 128), 256, 0, stream>>>(Xd_bf, WvT, bv, sc, out);
}
